// Embedding_5454608466258
// MI455X (gfx1250) — compile-verified, matrix operand order checked by reading, not run on hardware
//
#include <hip/hip_runtime.h>
#include <stdint.h>

// Problem constants (from reference: B=4096, VOCAB=50000, NUM_SPEC=2, EMBED=128)
#define K_A      49998      // effective K (vocab - special chars) = input row stride
#define N_E      128        // embed
#define M_B      4096       // batch
#define KT_N     1563       // ceil(K_A / 32) K-steps of 32
#define KT_FULL  1562       // steps with all 32 k-values valid
#define BLOB     1024       // bytes per (kt, nt) B fragment blob (32x16 bf16)
#define TILE_BYTES 8192     // 8 nt blobs per kt  (32 x 128 bf16)

typedef __bf16 bf16;
typedef bf16  v16bf __attribute__((ext_vector_type(16)));
typedef float v8f   __attribute__((ext_vector_type(8)));
typedef int   v4i   __attribute__((ext_vector_type(4)));

// Pointer types for the async-to-LDS builtin: (AS1 int4*, AS3 int4*, int, int)
typedef __attribute__((address_space(1))) v4i* gv4i_ptr;
typedef __attribute__((address_space(3))) v4i* lv4i_ptr;

// CDNA5 feature probes (graceful fallback if toolchain lacks them)
#if defined(__gfx1250__) && __has_builtin(__builtin_amdgcn_global_load_async_to_lds_b128) && \
    __has_builtin(__builtin_amdgcn_s_wait_asynccnt)
#define USE_ASYNC 1
#else
#define USE_ASYNC 0
#endif

// f32 -> bf16 bits, round-to-nearest-even (used in the one-shot swizzle kernel)
static __device__ __forceinline__ unsigned short f2bf_bits(float f) {
  union { float f; unsigned u; } x; x.f = f;
  unsigned r = x.u + 0x7FFFu + ((x.u >> 16) & 1u);
  return (unsigned short)(r >> 16);
}

// Pack two f32 -> bf16x2 in 3 VALU: 2x add (round-half-up) + v_perm_b32 splice
// of the two high halves. dst[15:0] = bf16(f0), dst[31:16] = bf16(f1).
static __device__ __forceinline__ unsigned pack2_bf16(float f0, float f1) {
  union { float f; unsigned u; } a, b; a.f = f0; b.f = f1;
  unsigned lo = a.u + 0x8000u;
  unsigned hi = b.u + 0x8000u;
  // byte sel: {s0.b3, s0.b2, s1.b3, s1.b2}  (s0 = hi, s1 = lo)
  return __builtin_amdgcn_perm(hi, lo, 0x07060302u);
}

// ---------------------------------------------------------------------------
// Kernel 1: swizzle word2vecs f32 -> bf16 into WMMA B-fragment layout.
// Blob for (kt, nt): bytes [0,512)   = lane l's first 8 bf16  (K = kb..kb+7)
//                    bytes [512,1024)= lane l's second 8 bf16 (K = kb+16..kb+23)
// where kb = kt*32 + ((l&16)?8:0), column n = nt*16 + (l&15).
// K zero-padded past 49998. One 256-thread block per kt.
// ---------------------------------------------------------------------------
__global__ void __launch_bounds__(256)
swz_kernel(const float* __restrict__ W, unsigned short* __restrict__ bswz) {
  int t = blockIdx.x * blockDim.x + threadIdx.x;
  if (t >= KT_N * 256) return;
  int lane = t & 31;
  int nt   = (t >> 5) & 7;
  int kt   = t >> 8;
  int n    = nt * 16 + (lane & 15);
  int kb   = kt * 32 + ((lane & 16) ? 8 : 0);
  size_t base = (size_t)(kt * 8 + nt) * 512 + (size_t)lane * 8;
#pragma unroll
  for (int j = 0; j < 16; ++j) {
    int k = kb + ((j < 8) ? j : (j + 8));   // j>=8 -> kb+16+(j-8)
    unsigned short v = 0;
    if (k < K_A) v = f2bf_bits(W[(size_t)k * N_E + n]);
    bswz[base + (size_t)(j >> 3) * 256 + (j & 7)] = v;
  }
}

// ---------------------------------------------------------------------------
// One K-step of 32: build A fragment (f32 -> bf16) and issue 8 WMMAs against
// the staged B tile at `ltile`.
// ---------------------------------------------------------------------------
static __device__ __forceinline__ void
compute_step(const unsigned char* __restrict__ ltile,
             const float* __restrict__ Arow,
             int kt, int koff, int lane, v8f acc[8])
{
  float av[16];
  const int kb = kt * 32 + koff;
  if (kt < KT_FULL) {
    const float2* __restrict__ p0 = (const float2*)(Arow + kb);      // 8B-aligned
    const float2* __restrict__ p1 = (const float2*)(Arow + kb + 16);
#pragma unroll
    for (int e = 0; e < 4; ++e) {
      float2 lo = p0[e], hi = p1[e];
      av[2 * e]     = lo.x; av[2 * e + 1]     = lo.y;
      av[8 + 2 * e] = hi.x; av[8 + 2 * e + 1] = hi.y;
    }
  } else {
    // K tail: only k < 49998 valid; rest contribute zero.
#pragma unroll
    for (int j = 0; j < 16; ++j) {
      int k = kb + ((j < 8) ? j : (j + 8));
      av[j] = (k < K_A) ? Arow[k] : 0.0f;
    }
  }

  union { unsigned u[8]; v16bf v; } af;
#pragma unroll
  for (int e = 0; e < 8; ++e)
    af.u[e] = pack2_bf16(av[2 * e], av[2 * e + 1]);

#pragma unroll
  for (int nt = 0; nt < 8; ++nt) {
    union { uint4 q[2]; v16bf v; } bf;
    bf.q[0] = *(const uint4*)(ltile + nt * BLOB +       (size_t)lane * 16);
    bf.q[1] = *(const uint4*)(ltile + nt * BLOB + 512 + (size_t)lane * 16);
    acc[nt] = __builtin_amdgcn_wmma_f32_16x16x32_bf16(
        false, af.v, false, bf.v, (short)0, acc[nt], false, false);
  }
}

// ---------------------------------------------------------------------------
// Kernel 2: GEMM. Block = 256 threads = 8 waves; each wave -> 16 rows x 128 cols.
// grid.x = M/128 blocks, grid.y = S K-chunks writing partials into ws.
// B tile staging: async-to-LDS double buffer (CDNA5) or sync copy fallback.
// ---------------------------------------------------------------------------
__global__ void __launch_bounds__(256)
gemm_kernel(const float* __restrict__ A,
            const unsigned char* __restrict__ bswz,
            float* __restrict__ part,
            int chunk_len)
{
#if USE_ASYNC
  __shared__ __align__(16) unsigned char ldsb[2 * TILE_BYTES];
#else
  __shared__ __align__(16) unsigned char ldsb[TILE_BYTES];
#endif

  const int tid  = threadIdx.x;
  const int lane = tid & 31;
  const int wave = tid >> 5;

  const int kt0 = blockIdx.y * chunk_len;
  const int kt1 = min(kt0 + chunk_len, KT_N);

  const int mbase = blockIdx.x * 128 + wave * 16;
  const int mrow  = mbase + (lane & 15);
  const float* __restrict__ Arow = A + (size_t)mrow * K_A;
  const int koff = (lane & 16) ? 8 : 0;

  v8f acc[8] = {};

#if USE_ASYNC
  // Each thread async-copies 32B of the 8KB tile (2 x b128), no VGPR data.
#define STAGE_ASYNC(KT, BUF)                                                      \
  do {                                                                            \
    const char* gsrc = (const char*)bswz + (size_t)(KT)*TILE_BYTES + (size_t)tid * 16; \
    char* ldst = (char*)ldsb + (size_t)(BUF)*TILE_BYTES + (size_t)tid * 16;       \
    __builtin_amdgcn_global_load_async_to_lds_b128(                               \
        (gv4i_ptr)gsrc, (lv4i_ptr)ldst, 0, 0);                                    \
    __builtin_amdgcn_global_load_async_to_lds_b128(                               \
        (gv4i_ptr)(gsrc + 4096), (lv4i_ptr)(ldst + 4096), 0, 0);                  \
  } while (0)

  if (kt0 < kt1) STAGE_ASYNC(kt0, 0);
  for (int kt = kt0; kt < kt1; ++kt) {
    const int cur = (kt - kt0) & 1;
    __builtin_amdgcn_s_wait_asynccnt(0);  // own async stores to LDS landed
    __syncthreads();                      // everyone's landed; prev compute done
    if (kt + 1 < kt1) STAGE_ASYNC(kt + 1, cur ^ 1);   // overlap with compute
    compute_step(ldsb + (size_t)cur * TILE_BYTES, Arow, kt, koff, lane, acc);
  }
#undef STAGE_ASYNC
#else
  for (int kt = kt0; kt < kt1; ++kt) {
    __syncthreads();   // protect LDS reuse across steps
    {
      const uint4* __restrict__ g = (const uint4*)(bswz + (size_t)kt * TILE_BYTES);
      uint4* l = (uint4*)ldsb;
      l[tid]       = g[tid];
      l[tid + 256] = g[tid + 256];
      if (kt + 1 < kt1)  // pull next tile toward the WGP (global_prefetch_b8)
        __builtin_prefetch(bswz + (size_t)(kt + 1) * TILE_BYTES + (size_t)tid * 32, 0, 1);
    }
    __syncthreads();
    compute_step(ldsb, Arow, kt, koff, lane, acc);
  }
#endif

  // ---- write partial tile (ISA C/D layout: VGPR v, lanes0-15 M=v, 16-31 M=v+8) ----
  float* __restrict__ P = part + (size_t)blockIdx.y * ((size_t)M_B * N_E);
  const int msel = (lane >> 4) & 1;
  const int ncol = lane & 15;
#pragma unroll
  for (int nt = 0; nt < 8; ++nt) {
#pragma unroll
    for (int v = 0; v < 8; ++v) {
      int m = mbase + msel * 8 + v;
      int n = nt * 16 + ncol;
      P[(size_t)m * N_E + n] = acc[nt][v];
    }
  }
}

// ---------------------------------------------------------------------------
// Kernel 3: reduce S partials into out.
// ---------------------------------------------------------------------------
__global__ void __launch_bounds__(256)
reduce_kernel(const float* __restrict__ part, float* __restrict__ out, int S) {
  int i = blockIdx.x * blockDim.x + threadIdx.x;
  if (i >= M_B * N_E) return;
  float s = 0.0f;
  for (int c = 0; c < S; ++c) s += part[(size_t)c * (M_B * N_E) + i];
  out[i] = s;
}

// ---------------------------------------------------------------------------
extern "C" void kernel_launch(void* const* d_in, const int* in_sizes, int n_in,
                              void* d_out, int out_size, void* d_ws, size_t ws_size,
                              hipStream_t stream) {
  const float* A = (const float*)d_in[0];   // (4096, 49998) f32
  const float* W = (const float*)d_in[1];   // (50000, 128) f32
  float* out = (float*)d_out;               // (4096, 128) f32
  unsigned char* ws = (unsigned char*)d_ws;

  const size_t BSWZ = (size_t)KT_N * TILE_BYTES;             // 12,804,096 B
  const size_t PER  = (size_t)M_B * N_E * sizeof(float);     // 2 MB per partial
  int S = 8;
  while (S > 1 && BSWZ + (size_t)S * PER > ws_size) --S;     // fit scratch

  // 1) bf16-swizzle the weight table into d_ws
  swz_kernel<<<KT_N, 256, 0, stream>>>(W, (unsigned short*)ws);

  // 2) split-K WMMA GEMM writing S partials
  int chunk = (KT_N + S - 1) / S;
  dim3 grid(M_B / 128, S);
  gemm_kernel<<<grid, 256, 0, stream>>>(A, ws, (float*)(ws + BSWZ), chunk);

  // 3) reduce partials -> out
  reduce_kernel<<<(M_B * N_E + 255) / 256, 256, 0, stream>>>(
      (const float*)(ws + BSWZ), out, S);

  (void)in_sizes; (void)n_in; (void)out_size;
}
